// DeepSeekV3MoE_45947560133086
// MI455X (gfx1250) — matrix operand order repaired and checked by reading, not compiled, not run on hardware
//
#include <hip/hip_runtime.h>
#include <hip/hip_bf16.h>
#include <math.h>

#define T_TOK   2048
#define HDIM    2048
#define IDIM    1024
#define NEXP    32
#define NGRP    4
#define GSIZE   8
#define TOPK    4
#define SCALE_F 2.5f
#define PAD_M   64             // expert token lists padded to 64-row tiles
#define CAP_ROWS 10240         // T*TOPK (8192) + worst-case pad (32*63=2016) rounded up
#define LDA     48             // LDS row stride (bf16 elems), keeps 16B chunk alignment
#define TILE_B  (64 * LDA * 2) // bytes per LDS tile buffer (6144)

typedef __attribute__((ext_vector_type(16))) __bf16 v16bf;
typedef __attribute__((ext_vector_type(8)))  __bf16 bf16x8;
typedef __attribute__((ext_vector_type(4)))  __bf16 bf16x4;
typedef __attribute__((ext_vector_type(8)))  float  v8f;

__device__ __forceinline__ float fast_sigmoid(float z) {
    return __builtin_amdgcn_rcpf(1.f + __expf(-z));   // v_rcp_f32, not IEEE div chain
}

// ---------------------------------------------------------------- fp32 -> bf16
__global__ void moe_cvt_bf16(const float* __restrict__ x, __bf16* __restrict__ y, int n) {
    int i = (blockIdx.x * blockDim.x + threadIdx.x) * 4;
    if (i + 3 < n) {
        float4 v = *(const float4*)(x + i);
        bf16x4 o;
        o[0] = (__bf16)v.x; o[1] = (__bf16)v.y; o[2] = (__bf16)v.z; o[3] = (__bf16)v.w;
        *(bf16x4*)(y + i) = o;
    }
}

// ---------------------------------------------------------------- gating (noaux_tc)
__global__ __launch_bounds__(32)
void moe_gate(const float* __restrict__ x, const float* __restrict__ gw,
              const float* __restrict__ bias,
              int* __restrict__ topk_idx, float* __restrict__ topk_w) {
    __shared__ float xs[HDIM];
    __shared__ float sc[NEXP];
    const int t = blockIdx.x;
    const int lane = threadIdx.x;

    for (int k = lane; k < HDIM; k += 32) xs[k] = x[(size_t)t * HDIM + k];
    __syncthreads();

    for (int e = 0; e < NEXP; ++e) {
        float p = 0.f;
        const float* w = gw + (size_t)e * HDIM;
        for (int k = lane; k < HDIM; k += 32) p += xs[k] * w[k];
        for (int o = 16; o > 0; o >>= 1) p += __shfl_down(p, o, 32);
        if (lane == 0) sc[e] = p;
    }
    __syncthreads();

    if (lane == 0) {
        float s[NEXP], sfc[NEXP];
        for (int e = 0; e < NEXP; ++e) {
            s[e]   = fast_sigmoid(sc[e]);
            sfc[e] = s[e] + bias[e];
        }
        float gs[NGRP];
        for (int g = 0; g < NGRP; ++g) {             // top-2 sum per group
            float m1 = -1e30f, m2 = -1e30f;
            for (int j = 0; j < GSIZE; ++j) {
                float v = sfc[g * GSIZE + j];
                if (v > m1) { m2 = m1; m1 = v; } else if (v > m2) m2 = v;
            }
            gs[g] = m1 + m2;
        }
        int g1 = 0;
        for (int g = 1; g < NGRP; ++g) if (gs[g] > gs[g1]) g1 = g;
        int g2 = -1;
        for (int g = 0; g < NGRP; ++g) if (g != g1 && (g2 < 0 || gs[g] > gs[g2])) g2 = g;

        float masked[NEXP];
        for (int e = 0; e < NEXP; ++e) {
            int g = e / GSIZE;
            masked[e] = (g == g1 || g == g2) ? sfc[e] : -1e30f;
        }
        int   idx[TOPK];
        float wsum = 0.f;
        for (int j = 0; j < TOPK; ++j) {
            int best = 0;
            for (int e = 1; e < NEXP; ++e) if (masked[e] > masked[best]) best = e;
            idx[j] = best;
            masked[best] = -2e30f;
            wsum += s[best];
        }
        float wnorm = __builtin_amdgcn_rcpf(wsum + 1e-20f) * SCALE_F;
        for (int j = 0; j < TOPK; ++j) {
            topk_idx[t * TOPK + j] = idx[j];
            topk_w  [t * TOPK + j] = s[idx[j]] * wnorm;
        }
    }
}

// ---------------------------------------------------------------- dispatch build
__global__ void moe_zero(int* cnt, int* cnt2) {
    int i = threadIdx.x;
    if (i < NEXP) { cnt[i] = 0; cnt2[i] = 0; }
}
__global__ void moe_count(const int* __restrict__ topk_idx, int* __restrict__ cnt) {
    int t = blockIdx.x * blockDim.x + threadIdx.x;
    if (t < T_TOK)
        for (int j = 0; j < TOPK; ++j) atomicAdd(&cnt[topk_idx[t * TOPK + j]], 1);
}
__global__ void moe_offsets(const int* __restrict__ cnt, int* __restrict__ off,
                            int* __restrict__ total) {
    if (threadIdx.x == 0) {
        int o = 0;
        for (int e = 0; e < NEXP; ++e) { off[e] = o; o += (cnt[e] + PAD_M - 1) & ~(PAD_M - 1); }
        off[NEXP] = o;
        *total = o;
    }
}
__global__ void moe_scatter(const int* __restrict__ topk_idx, const float* __restrict__ topk_w,
                            const int* __restrict__ off, int* __restrict__ cnt2,
                            int* __restrict__ tok_list, float* __restrict__ tok_w) {
    int t = blockIdx.x * blockDim.x + threadIdx.x;
    if (t < T_TOK) {
        for (int j = 0; j < TOPK; ++j) {
            int e = topk_idx[t * TOPK + j];
            int slot = atomicAdd(&cnt2[e], 1);
            int r = off[e] + slot;
            tok_list[r] = t;
            tok_w[r]   = topk_w[t * TOPK + j];
        }
    }
}
__global__ void moe_pad(const int* __restrict__ cnt, const int* __restrict__ off,
                        int* __restrict__ tok_list, float* __restrict__ tok_w,
                        int* __restrict__ row_expert) {
    int e = blockIdx.x;
    int c = cnt[e], p = (c + PAD_M - 1) & ~(PAD_M - 1), base = off[e];
    for (int s = c + (int)threadIdx.x; s < p; s += 64) { tok_list[base + s] = 0; tok_w[base + s] = 0.f; }
    for (int s = threadIdx.x; s < p; s += 64) row_expert[base + s] = e;
}

// ---------------------------------------------------------------- WMMA helpers
__device__ __forceinline__ v16bf frag_a(const __bf16* As, int mrow, int half) {
    const int kb1 = half * 8, kb2 = 16 + half * 8;         // ISA 16-bit A 16x32 layout
    bf16x8 lo = *(const bf16x8*)&As[mrow * LDA + kb1];
    bf16x8 hi = *(const bf16x8*)&As[mrow * LDA + kb2];
    v16bf f;
#pragma unroll
    for (int i = 0; i < 8; ++i) { f[i] = lo[i]; f[8 + i] = hi[i]; }
    return f;
}
__device__ __forceinline__ v16bf frag_b(const __bf16* Bs, int nrow, int half) {
    const int kb = half * 8;                               // lanes 0-15: K 0..7,16..23; 16-31: K 8..15,24..31
    bf16x8 lo = *(const bf16x8*)&Bs[nrow * LDA + kb];
    bf16x8 hi = *(const bf16x8*)&Bs[nrow * LDA + kb + 16];
    v16bf f;
#pragma unroll
    for (int i = 0; i < 8; ++i) { f[i] = lo[i]; f[8 + i] = hi[i]; }
    return f;
}
__device__ __forceinline__ void cvt8(const float* p, __bf16* dst) {
    float4 f0 = *(const float4*)(p);
    float4 f1 = *(const float4*)(p + 4);
    bf16x8 b;
    b[0]=(__bf16)f0.x; b[1]=(__bf16)f0.y; b[2]=(__bf16)f0.z; b[3]=(__bf16)f0.w;
    b[4]=(__bf16)f1.x; b[5]=(__bf16)f1.y; b[6]=(__bf16)f1.z; b[7]=(__bf16)f1.w;
    *(bf16x8*)dst = b;
}
// gfx1250 async global->LDS copy (b128), tracked by ASYNCcnt
__device__ __forceinline__ void async_ld_b128(unsigned lds_off, const void* gaddr) {
    asm volatile("global_load_async_to_lds_b128 %0, %1, off"
                 :: "v"(lds_off), "v"((unsigned long long)(size_t)gaddr) : "memory");
}
__device__ __forceinline__ void wait_async0() {
    asm volatile("s_wait_asynccnt 0x0" ::: "memory");
}
#define WMMA_BF16(A, B, C) \
    __builtin_amdgcn_wmma_f32_16x16x32_bf16(false, (A), false, (B), (short)0, (C), false, false)

// ---------------------------------------------------------------- fused gate+up GEMM
// 64x64 block tile, 4 waves each owning a 32x32 quadrant (2x2 WMMA tiles),
// double-buffered LDS: tile k+1 loads issue before tile k's WMMAs.
__global__ __launch_bounds__(128)
void moe_gemm_gate_up(const __bf16* __restrict__ X,
                      const float* __restrict__ W1, const float* __restrict__ W3,
                      __bf16* __restrict__ act,
                      const int* __restrict__ tok_list,
                      const int* __restrict__ row_expert,
                      const int* __restrict__ total_rows) {
    const int r0 = blockIdx.x * 64;
    if (total_rows && r0 >= *total_rows) return;
    const int n0  = blockIdx.y * 64;
    const int tid = threadIdx.x;
    const int lane = tid & 31, wave = tid >> 5;
    const int wm = (wave >> 1) * 32, wn = (wave & 1) * 32;
    const int half = lane >> 4, l16 = lane & 15;

    size_t wbase = row_expert ? (size_t)row_expert[r0] * (size_t)IDIM * HDIM : 0;
    const float* w1p = W1 + wbase;
    const float* w3p = W3 + wbase;

    __shared__ __align__(16) __bf16 As [2][64 * LDA];
    __shared__ __align__(16) __bf16 B1s[2][64 * LDA];
    __shared__ __align__(16) __bf16 B3s[2][64 * LDA];

    const int lrow = tid >> 1;            // 0..63
    const int lk   = (tid & 1) * 16;      // 0 or 16 (each thread owns 16 K-elems of its row)
    const int tok  = tok_list ? tok_list[r0 + lrow] : (r0 + lrow);
    const __bf16* aptr  = X   + (size_t)tok * HDIM + lk;
    const float*  b1ptr = w1p + (size_t)(n0 + lrow) * HDIM + lk;
    const float*  b3ptr = w3p + (size_t)(n0 + lrow) * HDIM + lk;
    const unsigned lds_a = (unsigned)(size_t)&As[0][lrow * LDA + lk];
    const int ldst = lrow * LDA + lk;

    auto load_tile = [&](int k0, int buf) {
        async_ld_b128(lds_a + buf * TILE_B,      aptr + k0);
        async_ld_b128(lds_a + buf * TILE_B + 16, aptr + k0 + 8);
        cvt8(b1ptr + k0,     &B1s[buf][ldst]);
        cvt8(b1ptr + k0 + 8, &B1s[buf][ldst + 8]);
        cvt8(b3ptr + k0,     &B3s[buf][ldst]);
        cvt8(b3ptr + k0 + 8, &B3s[buf][ldst + 8]);
    };

    v8f acc1[2][2] = {}; v8f acc3[2][2] = {};

    load_tile(0, 0);
    for (int k0 = 0; k0 < HDIM; k0 += 32) {
        const int cur = (k0 >> 5) & 1;
        wait_async0();
        __syncthreads();                       // tile `cur` resident in LDS
        if (k0 + 32 < HDIM) {
            load_tile(k0 + 32, cur ^ 1);       // pipeline next tile behind the WMMAs
            __builtin_prefetch(b1ptr + k0 + 64, 0, 1);
            __builtin_prefetch(b3ptr + k0 + 64, 0, 1);
        }
        const __bf16* Ac  = As [cur];
        const __bf16* B1c = B1s[cur];
        const __bf16* B3c = B3s[cur];
        v16bf a0 = frag_a(Ac, wm + l16,      half);
        v16bf a1 = frag_a(Ac, wm + 16 + l16, half);
#pragma unroll
        for (int j = 0; j < 2; ++j) {
            v16bf b1 = frag_b(B1c, wn + j * 16 + l16, half);
            acc1[0][j] = WMMA_BF16(a0, b1, acc1[0][j]);
            acc1[1][j] = WMMA_BF16(a1, b1, acc1[1][j]);
            v16bf b3 = frag_b(B3c, wn + j * 16 + l16, half);
            acc3[0][j] = WMMA_BF16(a0, b3, acc3[0][j]);
            acc3[1][j] = WMMA_BF16(a1, b3, acc3[1][j]);
        }
        __syncthreads();                       // reads of tile `cur` done before its reuse
    }
#pragma unroll
    for (int i = 0; i < 2; ++i)
#pragma unroll
        for (int j = 0; j < 2; ++j)
#pragma unroll
            for (int r = 0; r < 8; ++r) {      // C layout: M = r + 8*half, N = lane&15
                int row = r0 + wm + i * 16 + r + 8 * half;
                int col = n0 + wn + j * 16 + l16;
                float h1 = acc1[i][j][r], h3 = acc3[i][j][r];
                float v  = h1 * fast_sigmoid(h1) * h3;     // silu(h1)*h3
                act[(size_t)row * IDIM + col] = (__bf16)v;
            }
}

// ---------------------------------------------------------------- down GEMM
__global__ __launch_bounds__(128)
void moe_gemm_down(const __bf16* __restrict__ A,
                   const float* __restrict__ W2,
                   float* __restrict__ out,
                   const int* __restrict__ tok_list, const float* __restrict__ tok_w,
                   const int* __restrict__ row_expert,
                   const int* __restrict__ total_rows) {
    const int r0 = blockIdx.x * 64;
    if (total_rows && r0 >= *total_rows) return;
    const int n0  = blockIdx.y * 64;
    const int tid = threadIdx.x;
    const int lane = tid & 31, wave = tid >> 5;
    const int wm = (wave >> 1) * 32, wn = (wave & 1) * 32;
    const int half = lane >> 4, l16 = lane & 15;

    size_t wbase = row_expert ? (size_t)row_expert[r0] * (size_t)HDIM * IDIM : 0;
    const float* w2p = W2 + wbase;

    __shared__ __align__(16) __bf16 As[2][64 * LDA];
    __shared__ __align__(16) __bf16 Bs[2][64 * LDA];
    __shared__ int   s_tok[64];
    __shared__ float s_w[64];

    if (tid < 64) {
        int r = r0 + tid;
        s_tok[tid] = tok_list ? tok_list[r] : r;
        s_w[tid]   = tok_w ? tok_w[r] : 1.f;
    }

    const int lrow = tid >> 1;
    const int lk   = (tid & 1) * 16;
    const __bf16* aptr = A   + (size_t)(r0 + lrow) * IDIM + lk;
    const float*  bptr = w2p + (size_t)(n0 + lrow) * IDIM + lk;
    const unsigned lds_a = (unsigned)(size_t)&As[0][lrow * LDA + lk];
    const int ldst = lrow * LDA + lk;

    auto load_tile = [&](int k0, int buf) {
        async_ld_b128(lds_a + buf * TILE_B,      aptr + k0);
        async_ld_b128(lds_a + buf * TILE_B + 16, aptr + k0 + 8);
        cvt8(bptr + k0,     &Bs[buf][ldst]);
        cvt8(bptr + k0 + 8, &Bs[buf][ldst + 8]);
    };

    v8f acc[2][2] = {};
    load_tile(0, 0);
    for (int k0 = 0; k0 < IDIM; k0 += 32) {
        const int cur = (k0 >> 5) & 1;
        wait_async0();
        __syncthreads();
        if (k0 + 32 < IDIM) {
            load_tile(k0 + 32, cur ^ 1);
            __builtin_prefetch(bptr + k0 + 64, 0, 1);
        }
        const __bf16* Ac = As[cur];
        const __bf16* Bc = Bs[cur];
        v16bf a0 = frag_a(Ac, wm + l16,      half);
        v16bf a1 = frag_a(Ac, wm + 16 + l16, half);
#pragma unroll
        for (int j = 0; j < 2; ++j) {
            v16bf b = frag_b(Bc, wn + j * 16 + l16, half);
            acc[0][j] = WMMA_BF16(a0, b, acc[0][j]);
            acc[1][j] = WMMA_BF16(a1, b, acc[1][j]);
        }
        __syncthreads();
    }
#pragma unroll
    for (int i = 0; i < 2; ++i)
#pragma unroll
        for (int j = 0; j < 2; ++j)
#pragma unroll
            for (int r = 0; r < 8; ++r) {
                int m   = wm + i * 16 + r + 8 * half;
                int col = n0 + wn + j * 16 + l16;
                float v = acc[i][j][r];
                size_t oidx = (size_t)s_tok[m] * HDIM + col;
                if (tok_w) atomicAdd(&out[oidx], s_w[m] * v);
                else       out[oidx] = v;
            }
}

// ---------------------------------------------------------------- host launcher
extern "C" void kernel_launch(void* const* d_in, const int* in_sizes, int n_in,
                              void* d_out, int out_size, void* d_ws, size_t ws_size,
                              hipStream_t stream) {
    const float* x    = (const float*)d_in[0];
    const float* gw   = (const float*)d_in[1];
    const float* bias = (const float*)d_in[2];
    const float* w1   = (const float*)d_in[3];
    const float* w3   = (const float*)d_in[4];
    const float* w2   = (const float*)d_in[5];
    const float* ws1  = (const float*)d_in[6];
    const float* ws3  = (const float*)d_in[7];
    const float* ws2  = (const float*)d_in[8];
    float* out = (float*)d_out;

    char* ws = (char*)d_ws;
    size_t o = 0;
    auto take = [&](size_t bytes) -> void* {
        o = (o + 255) & ~(size_t)255;
        void* p = ws + o;
        o += bytes;
        return p;
    };
    __bf16* xb      = (__bf16*)take((size_t)T_TOK * HDIM * 2);
    __bf16* act_sh  = (__bf16*)take((size_t)T_TOK * IDIM * 2);
    __bf16* act_rt  = (__bf16*)take((size_t)CAP_ROWS * IDIM * 2);
    int*    tki     = (int*)   take((size_t)T_TOK * TOPK * 4);
    float*  tkw     = (float*) take((size_t)T_TOK * TOPK * 4);
    int*    cnt     = (int*)   take(NEXP * 4);
    int*    cnt2    = (int*)   take(NEXP * 4);
    int*    off     = (int*)   take((NEXP + 1) * 4);
    int*    total   = (int*)   take(4);
    int*    tok_lst = (int*)   take((size_t)CAP_ROWS * 4);
    float*  tok_w   = (float*) take((size_t)CAP_ROWS * 4);
    int*    row_exp = (int*)   take((size_t)CAP_ROWS * 4);
    (void)ws_size; (void)in_sizes; (void)n_in; (void)out_size;

    moe_cvt_bf16<<<(T_TOK * HDIM) / (256 * 4), 256, 0, stream>>>(x, xb, T_TOK * HDIM);
    moe_gate<<<T_TOK, 32, 0, stream>>>(x, gw, bias, tki, tkw);
    moe_zero<<<1, 64, 0, stream>>>(cnt, cnt2);
    moe_count<<<(T_TOK + 255) / 256, 256, 0, stream>>>(tki, cnt);
    moe_offsets<<<1, 32, 0, stream>>>(cnt, off, total);
    moe_scatter<<<(T_TOK + 255) / 256, 256, 0, stream>>>(tki, tkw, off, cnt2, tok_lst, tok_w);
    moe_pad<<<NEXP, 64, 0, stream>>>(cnt, off, tok_lst, tok_w, row_exp);

    dim3 blk(128);
    // gate+up GEMMs (shared expert, then routed)
    moe_gemm_gate_up<<<dim3(T_TOK / 64, IDIM / 64), blk, 0, stream>>>(
        xb, ws1, ws3, act_sh, nullptr, nullptr, nullptr);
    moe_gemm_gate_up<<<dim3(CAP_ROWS / 64, IDIM / 64), blk, 0, stream>>>(
        xb, w1, w3, act_rt, tok_lst, row_exp, total);

    // down GEMMs: shared writes (initializes out), routed atomically accumulates
    moe_gemm_down<<<dim3(T_TOK / 64, HDIM / 64), blk, 0, stream>>>(
        act_sh, ws2, out, nullptr, nullptr, nullptr, nullptr);
    moe_gemm_down<<<dim3(CAP_ROWS / 64, HDIM / 64), blk, 0, stream>>>(
        act_rt, w2, out, tok_lst, tok_w, row_exp, total);
}
